// Loss_4655744549122
// MI455X (gfx1250) — matrix operand order repaired and checked by reading, not compiled, not run on hardware
//
#include <hip/hip_runtime.h>
#include <hip/hip_bf16.h>
#include <math.h>

typedef __attribute__((ext_vector_type(16))) _Float16 v16h;
typedef __attribute__((ext_vector_type(8)))  float    v8f;

#define N_IMG 128
#define N_TXT 128
#define RREG  48
#define LWRD  64
#define DIM   1024
#define EPSV  1e-8f
#define LAMB  20.0f

// ---------------------------------------------------------------------------
// WMMA fragment loaders for V_WMMA_F32_16X16X32_F16 (wave32), per ISA layouts.
// A: 16x32 f16 row-major. lane<16: row=M=lane, K = {k0..k0+7, k0+16..k0+23}
//                         lane>=16: row=M=lane-16, K = {k0+8..+15, k0+24..+31}
// B: given as BT row-major [N x K]. lane<16: col=N=lane, K = {k0..k0+7, k0+16..k0+23}
//                                   lane>=16: col=lane-16, K = {k0+8..+15, k0+24..+31}
// ---------------------------------------------------------------------------
__device__ __forceinline__ v16h load_a_frag(const _Float16* A, int lda, int row0, int k0) {
  int lane = threadIdx.x & 31;
  const _Float16* p = A + (size_t)(row0 + (lane & 15)) * lda + k0 + ((lane < 16) ? 0 : 8);
  union { v16h v; uint4 q[2]; } u;
  u.q[0] = *(const uint4*)(p);
  u.q[1] = *(const uint4*)(p + 16);
  return u.v;
}
__device__ __forceinline__ v16h load_b_frag(const _Float16* BT, int ldb, int col0, int k0) {
  int lane = threadIdx.x & 31;
  const _Float16* p = BT + (size_t)(col0 + (lane & 15)) * ldb + k0 + ((lane < 16) ? 0 : 8);
  union { v16h v; uint4 q[2]; } u;
  u.q[0] = *(const uint4*)(p);
  u.q[1] = *(const uint4*)(p + 16);
  return u.v;
}
#define WMMA_F32(a, b, c) \
  __builtin_amdgcn_wmma_f32_16x16x32_f16(false, (a), false, (b), (short)0, (c), false, false)

// async copy of one 16B chunk: global -> LDS, ASYNCcnt-tracked (gfx1250)
__device__ __forceinline__ void async_g2l_b128(unsigned lds_byte_addr, const void* gaddr) {
  asm volatile("global_load_async_to_lds_b128 %0, %1, off"
               :: "v"(lds_byte_addr), "v"(gaddr) : "memory");
}

// ---------------------------------------------------------------------------
// prep: l2norm img_q -> qn16 (f16); img_v -> iv16 (f16) + |img_v| (f32)
// ---------------------------------------------------------------------------
__global__ void prep_img_kernel(const float* __restrict__ img_q, const float* __restrict__ img_v,
                                _Float16* __restrict__ qn16, _Float16* __restrict__ iv16,
                                float* __restrict__ ivnorm) {
  int row = blockIdx.x;                 // n*RREG + r
  const float* q = img_q + (size_t)row * DIM;
  const float* v = img_v + (size_t)row * DIM;
  __shared__ float red[256];
  int tid = threadIdx.x;
  float qv[4], vv[4], sq = 0.f, sv = 0.f;
  for (int i = 0; i < 4; ++i) {
    qv[i] = q[tid + 256 * i]; vv[i] = v[tid + 256 * i];
    sq += qv[i] * qv[i]; sv += vv[i] * vv[i];
  }
  red[tid] = sq; __syncthreads();
  for (int s = 128; s > 0; s >>= 1) { if (tid < s) red[tid] += red[tid + s]; __syncthreads(); }
  float nq = sqrtf(red[0]) + EPSV;
  __syncthreads();
  red[tid] = sv; __syncthreads();
  for (int s = 128; s > 0; s >>= 1) { if (tid < s) red[tid] += red[tid + s]; __syncthreads(); }
  float nv = sqrtf(red[0]);
  for (int i = 0; i < 4; ++i) {
    qn16[(size_t)row * DIM + tid + 256 * i] = (_Float16)(qv[i] / nq);
    iv16[(size_t)row * DIM + tid + 256 * i] = (_Float16)vv[i];
  }
  if (tid == 0) ivnorm[row] = nv;
}

// mask padded words, l2norm txt_k -> tk16; masked txt_v -> tv16 + |tv| (f32)
__global__ void prep_txt_kernel(const float* __restrict__ txt_k, const float* __restrict__ txt_v,
                                const int* __restrict__ tlen,
                                _Float16* __restrict__ tk16, _Float16* __restrict__ tv16,
                                float* __restrict__ tvnorm) {
  int row = blockIdx.x;                 // t*LWRD + l
  int t = row >> 6, l = row & 63;
  float m = (l < tlen[t]) ? 1.f : 0.f;
  const float* kp = txt_k + (size_t)row * DIM;
  const float* vp = txt_v + (size_t)row * DIM;
  __shared__ float red[256];
  int tid = threadIdx.x;
  float kv[4], vv[4], sk = 0.f, sv = 0.f;
  for (int i = 0; i < 4; ++i) {
    kv[i] = kp[tid + 256 * i] * m; vv[i] = vp[tid + 256 * i] * m;
    sk += kv[i] * kv[i]; sv += vv[i] * vv[i];
  }
  red[tid] = sk; __syncthreads();
  for (int s = 128; s > 0; s >>= 1) { if (tid < s) red[tid] += red[tid + s]; __syncthreads(); }
  float nk = sqrtf(red[0]) + EPSV;
  __syncthreads();
  red[tid] = sv; __syncthreads();
  for (int s = 128; s > 0; s >>= 1) { if (tid < s) red[tid] += red[tid + s]; __syncthreads(); }
  float nv = sqrtf(red[0]);
  for (int i = 0; i < 4; ++i) {
    tk16[(size_t)row * DIM + tid + 256 * i] = (_Float16)(kv[i] / nk);
    tv16[(size_t)row * DIM + tid + 256 * i] = (_Float16)vv[i];
  }
  if (tid == 0) tvnorm[row] = nv;
}

// ---------------------------------------------------------------------------
// Gram:  G[n] = X[n] @ X[n]^T  over d=1024, one wave per 16x16 tile (WMMA f16)
// two interleaved accumulators to break the WMMA->WMMA C-chain
// ---------------------------------------------------------------------------
__global__ void gram_kernel(const _Float16* __restrict__ X, float* __restrict__ G,
                            int rows, int tpd) {
  int n = blockIdx.x;
  int tile = blockIdx.y;
  int ti = tile / tpd, tj = tile % tpd;
  const _Float16* Xn = X + (size_t)n * rows * DIM;
  v8f accA = {}, accB = {};
  for (int k = 0; k < DIM; k += 64) {
    v16h a0 = load_a_frag(Xn, DIM, 16 * ti, k);
    v16h b0 = load_b_frag(Xn, DIM, 16 * tj, k);
    v16h a1 = load_a_frag(Xn, DIM, 16 * ti, k + 32);
    v16h b1 = load_b_frag(Xn, DIM, 16 * tj, k + 32);
    accA = WMMA_F32(a0, b0, accA);
    accB = WMMA_F32(a1, b1, accB);
  }
  v8f acc = accA + accB;
  int lane = threadIdx.x & 31;
  int col = lane & 15, rb = (lane < 16) ? 0 : 8;
  float* Gn = G + (size_t)n * rows * rows;
#pragma unroll
  for (int v = 0; v < 8; ++v)
    Gn[(size_t)(16 * ti + rb + v) * rows + 16 * tj + col] = acc[v];
}

// ---------------------------------------------------------------------------
// Main per-pair kernel. LDS pool partitions (floats):
//   A1  [48][65] @0       i2t attention a'
//   A2  [64][49] @3120    t2i attention a2'
//   X   @6256 (6848 floats):
//     phases A..B2 : Sm[64][49] @X, Gm[48][65] @X+3136
//     phase  B3    : GttS[64][68] @X (16B-aligned rows for async b128),
//                    GiiS[48][52] @X+4352
// ---------------------------------------------------------------------------
#define POOL_F 13104
#define A1v(r, l)   pool[(r) * 65 + (l)]
#define A2v(l, r)   pool[3120 + (l) * 49 + (r)]
#define smS(l, r)   pool[6256 + (l) * 49 + (r)]
#define Gmv(r, l)   pool[6256 + 3136 + (r) * 65 + (l)]
#define GTT_OFF     6256
#define GII_OFF     (6256 + 4352)
#define GttSv(a, b) pool[GTT_OFF + (a) * 68 + (b)]
#define GiiSv(a, b) pool[GII_OFF + (a) * 52 + (b)]

__global__ __launch_bounds__(256) void pair_kernel(
    const _Float16* __restrict__ qn16, const _Float16* __restrict__ iv16,
    const _Float16* __restrict__ tk16, const _Float16* __restrict__ tv16,
    const float* __restrict__ ivnorm, const float* __restrict__ tvnorm,
    const float* __restrict__ Gii, const float* __restrict__ Gtt,
    const int* __restrict__ tlen, float* __restrict__ out) {
  int n = blockIdx.x, t = blockIdx.y;
  const _Float16* qn = qn16 + (size_t)n * RREG * DIM;
  const _Float16* iv = iv16 + (size_t)n * RREG * DIM;
  const _Float16* tk = tk16 + (size_t)t * LWRD * DIM;
  const _Float16* tv = tv16 + (size_t)t * LWRD * DIM;
  const float* Gii_n = Gii + (size_t)n * RREG * RREG;
  const float* Gtt_t = Gtt + (size_t)t * LWRD * LWRD;
  const float* ivn_n = ivnorm + (size_t)n * RREG;
  const float* tvn_t = tvnorm + (size_t)t * LWRD;
  int n_word = tlen[t];

  __shared__ float pool[POOL_F];
  __shared__ float rnL[LWRD], rn2R[RREG];
  __shared__ float q1[RREG], w12a[RREG], q2[LWRD], w12b[LWRD];
  __shared__ float accI, accT;

  int tid = threadIdx.x, wave = tid >> 5, lane = tid & 31;

  // warm the Gram rows into cache while WMMA runs (gfx1250 global_prefetch_b8)
  for (int ln = tid; ln < 128; ln += 256) __builtin_prefetch(Gtt_t + ln * 32, 0, 0);
  for (int ln = tid; ln < 72;  ln += 256) __builtin_prefetch(Gii_n + ln * 32, 0, 0);

  // ---- Phase A: S[64,48] = tk . qn^T ; G[48,64] = iv . tv^T (K = 1024) ----
  v8f acc0 = {}, acc1 = {}, acc2 = {};
  if (wave < 4) {
    int l0 = 16 * wave;
    for (int k = 0; k < DIM; k += 32) {
      v16h a  = load_a_frag(tk, DIM, l0, k);
      v16h b0 = load_b_frag(qn, DIM, 0,  k);
      v16h b1 = load_b_frag(qn, DIM, 16, k);
      v16h b2 = load_b_frag(qn, DIM, 32, k);
      acc0 = WMMA_F32(a, b0, acc0);
      acc1 = WMMA_F32(a, b1, acc1);
      acc2 = WMMA_F32(a, b2, acc2);
    }
    int col = lane & 15, rb = (lane < 16) ? 0 : 8;
#pragma unroll
    for (int v = 0; v < 8; ++v) {
      smS(l0 + rb + v, col)      = acc0[v];
      smS(l0 + rb + v, 16 + col) = acc1[v];
      smS(l0 + rb + v, 32 + col) = acc2[v];
    }
  } else {
    int c0 = 16 * (wave - 4);
    for (int k = 0; k < DIM; k += 32) {
      v16h b  = load_b_frag(tv, DIM, c0, k);
      v16h a0 = load_a_frag(iv, DIM, 0,  k);
      v16h a1 = load_a_frag(iv, DIM, 16, k);
      v16h a2 = load_a_frag(iv, DIM, 32, k);
      acc0 = WMMA_F32(a0, b, acc0);
      acc1 = WMMA_F32(a1, b, acc1);
      acc2 = WMMA_F32(a2, b, acc2);
    }
    int col = lane & 15, rb = (lane < 16) ? 0 : 8;
#pragma unroll
    for (int v = 0; v < 8; ++v) {
      Gmv(rb + v,      c0 + col) = acc0[v];
      Gmv(16 + rb + v, c0 + col) = acc1[v];
      Gmv(32 + rb + v, c0 + col) = acc2[v];
    }
  }
  __syncthreads();

  // ---- Phase B1: E = leaky_relu(S) in place; zero accumulators ----
  for (int idx = tid; idx < LWRD * RREG; idx += 256) {
    int l = idx / RREG, r = idx - l * RREG;
    float x = smS(l, r);
    smS(l, r) = (x > 0.f) ? x : 0.1f * x;
  }
  if (tid < RREG) q1[tid] = 0.f;
  if (tid >= 128 && tid < 128 + LWRD) q2[tid - 128] = 0.f;
  if (tid == 255) { accI = 0.f; accT = 0.f; }
  __syncthreads();

  // row norms: rnL[l] over r (i2t), rn2R[r] over l (t2i)
  if (tid < LWRD) {
    float s = 0.f;
    for (int r = 0; r < RREG; ++r) { float e = smS(tid, r); s += e * e; }
    rnL[tid] = sqrtf(s) + EPSV;
  } else if (tid < LWRD + RREG) {
    int r = tid - LWRD;
    float s = 0.f;
    for (int l = 0; l < LWRD; ++l) { float e = smS(l, r); s += e * e; }
    rn2R[r] = sqrtf(s) + EPSV;
  }
  __syncthreads();

  // ---- Phase B2: softmax + focal renorm + cosine numerators ----
  if (tid < RREG) {                      // i2t: one thread per region r
    int r = tid;
    float nw = (float)n_word;
    float mx = -1e30f;
    for (int l = 0; l < LWRD; ++l) {
      float x = (l < n_word) ? LAMB * smS(l, r) / rnL[l] : -1e30f;
      mx = fmaxf(mx, x);
    }
    float se = 0.f;
    for (int l = 0; l < LWRD; ++l) {
      float x = (l < n_word) ? LAMB * smS(l, r) / rnL[l] : -1e30f;
      se += expf(x - mx);
    }
    float inv_se = 1.f / se, sa = 0.f;
    for (int l = 0; l < LWRD; ++l) {
      float x = (l < n_word) ? LAMB * smS(l, r) / rnL[l] : -1e30f;
      float p = expf(x - mx) * inv_se;
      A1v(r, l) = p; sa += p;
    }
    float ts = 0.f;
    for (int l = 0; l < LWRD; ++l) { float p = A1v(r, l); if (p * nw - sa > 0.f) ts += p; }
    float invts = 1.f / ((ts > 0.f) ? ts : 1.f);
    float w = 0.f;
    for (int l = 0; l < LWRD; ++l) {
      float p = A1v(r, l);
      float ap = (p * nw - sa > 0.f) ? p * invts : 0.f;
      A1v(r, l) = ap;
      w += ap * Gmv(r, l);
    }
    w12a[r] = w;
  } else if (tid < RREG + LWRD) {        // t2i: one thread per word l
    int l = tid - RREG;
    const float srcL = (float)RREG;
    float mx = -1e30f;
    for (int r = 0; r < RREG; ++r) mx = fmaxf(mx, LAMB * smS(l, r) / rn2R[r]);
    float se = 0.f;
    for (int r = 0; r < RREG; ++r) se += expf(LAMB * smS(l, r) / rn2R[r] - mx);
    float inv_se = 1.f / se, sa = 0.f;
    for (int r = 0; r < RREG; ++r) {
      float p = expf(LAMB * smS(l, r) / rn2R[r] - mx) * inv_se;
      A2v(l, r) = p; sa += p;
    }
    float ts = 0.f;
    for (int r = 0; r < RREG; ++r) { float p = A2v(l, r); if (p * srcL - sa > 0.f) ts += p; }
    float invts = 1.f / ((ts > 0.f) ? ts : 1.f);
    float w = 0.f;
    for (int r = 0; r < RREG; ++r) {
      float p = A2v(l, r);
      float ap = (p * srcL - sa > 0.f) ? p * invts : 0.f;
      A2v(l, r) = ap;
      w += ap * Gmv(r, l);
    }
    w12b[l] = w;
  }
  __syncthreads();

  // ---- Phase B2.5: async-stage Gtt/Gii into LDS over the dead S/G region ----
  {
    unsigned gtt_base = (unsigned)(size_t)(void*)&pool[GTT_OFF];
    unsigned gii_base = (unsigned)(size_t)(void*)&pool[GII_OFF];
    // Gtt: 64 rows x 16 chunks of 16B (row stride 68 floats = 272B, 16B aligned)
    for (int idx = tid; idx < 64 * 16; idx += 256) {
      int a = idx >> 4, c = idx & 15;
      async_g2l_b128(gtt_base + (unsigned)((a * 68 + c * 4) * 4), Gtt_t + a * 64 + c * 4);
    }
    // Gii: 48 rows x 12 chunks of 16B (row stride 52 floats = 208B, 16B aligned)
    for (int idx = tid; idx < 48 * 12; idx += 256) {
      int a = idx / 12, c = idx - a * 12;
      async_g2l_b128(gii_base + (unsigned)((a * 52 + c * 4) * 4), Gii_n + a * 48 + c * 4);
    }
    asm volatile("s_wait_asynccnt 0" ::: "memory");
  }
  __syncthreads();

  // ---- Phase B3: quadratic forms  |wei|^2 = a' G a'  (distributed MACs) ----
  for (int idx = tid; idx < RREG * LWRD; idx += 256) {
    int r = idx >> 6, l = idx & 63;
    float a = A1v(r, l);
    if (a != 0.f) {
      float u = 0.f;
      for (int l2 = 0; l2 < LWRD; ++l2) u += A1v(r, l2) * GttSv(l2, l);
      atomicAdd(&q1[r], a * u);
    }
  }
  for (int idx = tid; idx < LWRD * RREG; idx += 256) {
    int l = idx / RREG, r = idx - l * RREG;
    float a = A2v(l, r);
    if (a != 0.f) {
      float u = 0.f;
      for (int r2 = 0; r2 < RREG; ++r2) u += A2v(l, r2) * GiiSv(r2, r);
      atomicAdd(&q2[l], a * u);
    }
  }
  __syncthreads();

  // ---- Phase B4: cosines (with l2norm-eps semantics) + reductions ----
  if (tid < RREG) {
    int r = tid;
    float tq = sqrtf(fmaxf(q1[r], 0.f));
    float sc = 1.f / (tq + EPSV);
    float num = w12a[r] * sc;
    float den = fmaxf(ivn_n[r] * (tq * sc), EPSV);
    atomicAdd(&accI, num / den);
  } else if (tid < RREG + LWRD) {
    int l = tid - RREG;
    if (l < n_word) {
      float tq = sqrtf(fmaxf(q2[l], 0.f));
      float sc = 1.f / (tq + EPSV);
      float num = w12b[l] * sc;
      float den = fmaxf(tvn_t[l] * (tq * sc), EPSV);
      atomicAdd(&accT, num / den);
    }
  }
  __syncthreads();
  if (tid == 0) {
    out[(size_t)n * N_TXT + t] = accI / (float)RREG;
    out[(size_t)N_IMG * N_TXT + (size_t)n * N_TXT + t] = accT / (float)n_word;
  }
}

// ---------------------------------------------------------------------------
extern "C" void kernel_launch(void* const* d_in, const int* in_sizes, int n_in,
                              void* d_out, int out_size, void* d_ws, size_t ws_size,
                              hipStream_t stream) {
  (void)in_sizes; (void)n_in; (void)out_size; (void)ws_size;
  const float* img_q = (const float*)d_in[0];
  const float* img_v = (const float*)d_in[1];
  const float* txt_k = (const float*)d_in[2];
  const float* txt_v = (const float*)d_in[3];
  const int*   tlen  = (const int*)d_in[4];
  float* out = (float*)d_out;

  char* ws = (char*)d_ws;
  size_t off = 0;
  auto take = [&](size_t bytes) -> void* {
    void* p = ws + off;
    off += (bytes + 255) & ~(size_t)255;
    return p;
  };
  _Float16* qn16 = (_Float16*)take((size_t)N_IMG * RREG * DIM * 2);   // 12.6 MB
  _Float16* iv16 = (_Float16*)take((size_t)N_IMG * RREG * DIM * 2);   // 12.6 MB
  _Float16* tk16 = (_Float16*)take((size_t)N_TXT * LWRD * DIM * 2);   // 16.8 MB
  _Float16* tv16 = (_Float16*)take((size_t)N_TXT * LWRD * DIM * 2);   // 16.8 MB
  float* ivnorm  = (float*)take((size_t)N_IMG * RREG * 4);
  float* tvnorm  = (float*)take((size_t)N_TXT * LWRD * 4);
  float* Gii     = (float*)take((size_t)N_IMG * RREG * RREG * 4);     // 1.2 MB
  float* Gtt     = (float*)take((size_t)N_TXT * LWRD * LWRD * 4);     // 2.1 MB

  prep_img_kernel<<<N_IMG * RREG, 256, 0, stream>>>(img_q, img_v, qn16, iv16, ivnorm);
  prep_txt_kernel<<<N_TXT * LWRD, 256, 0, stream>>>(txt_k, txt_v, tlen, tk16, tv16, tvnorm);
  gram_kernel<<<dim3(N_IMG, 9), 32, 0, stream>>>(iv16, Gii, RREG, 3);
  gram_kernel<<<dim3(N_TXT, 16), 32, 0, stream>>>(tv16, Gtt, LWRD, 4);
  pair_kernel<<<dim3(N_IMG, N_TXT), 256, 0, stream>>>(qn16, iv16, tk16, tv16,
                                                      ivnorm, tvnorm, Gii, Gtt, tlen, out);
}
